// CounterModel_4105988735540
// MI455X (gfx1250) — matrix operand order repaired and checked by reading, not compiled, not run on hardware
//
#include <hip/hip_runtime.h>
#include <cstdint>

// ---------------------------------------------------------------------------
// 5-layer GRU, H=20, B=256, T=4096.
// Latency-bound recurrence -> per-step dependent chain minimized with
// v_wmma_f32_16x16x32_f16 (K=32 covers H=20 in ONE matrix op per gate tile).
// 16 blocks = 16 independent batch chunks of 16.
// 5 waves / block = 5 layers, diagonally software-pipelined over 4-step time
// tiles through double-buffered LDS interfaces (one s_barrier per tile).
// Activations: 3 transcendentals per element (v_tanh_f32 only).
// K axis is permuted (pi(2i)=i, pi(2i+1)=16+i) so a lane's two column-parts
// pack into one dword -> v_cvt_pk_rtz + single ds_store_b32 per row.
// ---------------------------------------------------------------------------

#define H_    20
#define HP_   32          // padded hidden (K and column pad for WMMA)
#define T_    4096
#define B_    256
#define L_    5
#define CHK_  16          // batch rows per block (= WMMA M)
#define TILE_ 4           // timesteps per pipeline phase
#define NT_   (T_ / TILE_)

typedef __attribute__((ext_vector_type(16))) _Float16 v16h;
typedef __attribute__((ext_vector_type(8)))  _Float16 v8h;
typedef __attribute__((ext_vector_type(2)))  __fp16   v2fp16;  // cvt_pkrtz ret
typedef __attribute__((ext_vector_type(8)))  float    v8f;

// LDS layout (halfs):
//   iface[l][buf][tt][b][HP_]  l in 0..3 : layer l -> l+1 handoff, double buffered
//   in0  [tt][b][HP_]          : layer-0 input staging (x at slot 0, rest 0)
//   hscr [wave][b][HP_]        : per-wave h D->A transpose scratch
// All K slots hold feature pi(slot); producers and weight fragments agree.
#define IF_STRIDE (TILE_ * CHK_ * HP_)                 // 2048 halfs
#define SM_IFACE  0
#define SM_IN0    ((L_ - 1) * 2 * IF_STRIDE)           // 16384
#define SM_HSCR   (SM_IN0 + IF_STRIDE)                 // 18432
#define SM_TOTAL  (SM_HSCR + L_ * CHK_ * HP_)          // 20992 halfs = 41 KB

struct GruParams {
  const float* x;
  const float* Wih[L_];
  const float* Whh[L_];
  const float* bih[L_];
  const float* bhh[L_];
  float* out_h;    // [B,T,H]
  float* out_hn;   // [B,L,H]
};

static __device__ __forceinline__ v8f wmma16x16x32(v16h a, v16h b, v8f c) {
  // D = A(16x32 f16) x B(32x16 f16) + C(16x16 f32)
  return __builtin_amdgcn_wmma_f32_16x16x32_f16(false, a, false, b, (short)0, c,
                                                false, false);
}

static __device__ __forceinline__ v8f splat8(float v) {
  v8f c;
#pragma unroll
  for (int i = 0; i < 8; ++i) c[i] = v;
  return c;
}

// Fast activations: exactly one transcendental per gate element.
static __device__ __forceinline__ float tanhfast(float x) {
#if __has_builtin(__builtin_amdgcn_tanhf)
  return __builtin_amdgcn_tanhf(x);        // gfx1250 v_tanh_f32 (confirmed)
#else
  return 1.0f - 2.0f * __builtin_amdgcn_rcpf(__expf(2.0f * x) + 1.0f);
#endif
}
static __device__ __forceinline__ float sigm(float x) {
  return __builtin_fmaf(0.5f, tanhfast(0.5f * x), 0.5f);  // 0.5*(1+tanh(x/2))
}

// A-matrix (16x32 f16) fragment load from a 32-half row (permuted K slots):
// lanes 0-15 (hi=0): e0-7 = slots 0-7,  e8-15 = slots 16-23
// lanes16-31 (hi=1): e0-7 = slots 8-15, e8-15 = slots 24-31
static __device__ __forceinline__ v16h loadA(const _Float16* row, int hi) {
  const v8h* p0 = (const v8h*)(row + (hi ? 8 : 0));
  const v8h* p1 = (const v8h*)(row + 16 + (hi ? 8 : 0));
  v16h a;
  ((v8h*)&a)[0] = *p0;
  ((v8h*)&a)[1] = *p1;
  return a;
}

// K-slot -> feature permutation: pi(2i)=i, pi(2i+1)=16+i.
static __device__ __forceinline__ int kperm(int k) {
  return (k & 1) ? 16 + (k >> 1) : (k >> 1);
}

__launch_bounds__(L_ * 32, 1)
__global__ void gru_pipeline(GruParams P) {
  __shared__ __align__(16) _Float16 smem[SM_TOTAL];

  const int tid  = threadIdx.x;
  const int lane = tid & 31;
  const int wave = tid >> 5;            // wave == layer index
  const int b0   = blockIdx.x * CHK_;   // batch chunk base

  // Zero all of LDS once: h state, K/column pads, interface pads stay 0 forever.
  {
    uint32_t* w = (uint32_t*)smem;
    for (int i = tid; i < SM_TOTAL / 2; i += L_ * 32) w[i] = 0u;
  }

  const int nCol  = lane & 15;          // N column within a 16-wide tile
  const int hiH   = lane >> 4;          // lane-half selector
  const int kbase = hiH ? 16 : 0;       // unused; B-fragment K slots start at 8*hiH
  const int in_sz = (wave == 0) ? 1 : H_;

  const float* Wih = P.Wih[wave];
  const float* Whh = P.Whh[wave];
  const float* bih = P.bih[wave];
  const float* bhh = P.bhh[wave];

  // Preload weights into B-layout (32x16 f16) fragments, K-permuted.
  // 6 tiles: gate g (r,z,n) x column part p (cols 0-15 / 16-31 of 32-padded gate).
  v16h Bi[6], Bh[6];
#pragma unroll
  for (int g = 0; g < 3; ++g) {
#pragma unroll
    for (int p = 0; p < 2; ++p) {
      const int  idx = g * 2 + p;
      const int  cg  = p * 16 + nCol;        // column within padded gate
      const bool cv  = cg < H_;              // real gate unit?
      const int  gc  = g * H_ + cg;          // row in W (gate-major)
      v16h bi, bh;
#pragma unroll
      for (int e = 0; e < 16; ++e) {
        const int k = (e & 7) + 8 * hiH + ((e >> 3) << 4);  // WMMA K slot
        const int f = kperm(k);              // feature held in that slot
        float wi = (cv && f < in_sz) ? Wih[gc * in_sz + f] : 0.0f;
        float wh = (cv && f < H_)    ? Whh[gc * H_   + f] : 0.0f;
        bi[e] = (_Float16)wi;
        bh[e] = (_Float16)wh;
      }
      Bi[idx] = bi;
      Bh[idx] = bh;
    }
  }

  // Per-lane fused biases. r/z fold b_ih+b_hh into the x-projection C operand;
  // n keeps b_ih_n (x side) and b_hh_n (h side) separate since hn is scaled by r.
  float brz[4], bxn[2], bhn[2];
#pragma unroll
  for (int p = 0; p < 2; ++p) {
    const int  c = p * 16 + nCol;
    const bool v = c < H_;
    brz[0 + p] = v ? (bih[0 * H_ + c] + bhh[0 * H_ + c]) : 0.0f;  // r
    brz[2 + p] = v ? (bih[1 * H_ + c] + bhh[1 * H_ + c]) : 0.0f;  // z
    bxn[p]     = v ? bih[2 * H_ + c] : 0.0f;                      // n (x side)
    bhn[p]     = v ? bhh[2 * H_ + c] : 0.0f;                      // n (h side)
  }

  // h state in D-layout f32: [col part p][row reg r] ; padded cols stay 0.
  float hst[2][8];
#pragma unroll
  for (int p = 0; p < 2; ++p)
#pragma unroll
    for (int r = 0; r < 8; ++r) hst[p][r] = 0.0f;

  _Float16* hs = &smem[SM_HSCR + wave * CHK_ * HP_];
  const bool last_layer = (wave == L_ - 1);

  // Diagonal pipeline: at phase ph, wave l works on time-tile (ph - l).
  for (int ph = 0; ph < NT_ + L_ - 1; ++ph) {
    __syncthreads();                       // all waves, every phase
    const int mt = ph - wave;
    if (mt < 0 || mt >= NT_) continue;
    const int bufsel = mt & 1;

    if (wave == 0) {
      // Stage x for this tile (feature 0 -> slot 0; pads already zero).
#pragma unroll
      for (int tq = 0; tq < 2; ++tq) {
        const int tt = hiH + 2 * tq;
        const int t  = mt * TILE_ + tt;
        const float xv = P.x[(size_t)(b0 + nCol) * T_ + t];
        smem[SM_IN0 + (tt * CHK_ + nCol) * HP_] = (_Float16)xv;
      }
    }

    const _Float16* inb =
        (wave == 0) ? &smem[SM_IN0]
                    : &smem[SM_IFACE + ((wave - 1) * 2 + bufsel) * IF_STRIDE];
    _Float16* outb =
        (!last_layer) ? &smem[SM_IFACE + (wave * 2 + bufsel) * IF_STRIDE]
                      : nullptr;

    for (int tt = 0; tt < TILE_; ++tt) {
      const int t = mt * TILE_ + tt;

      // Input projection (independent of h -> overlaps the recurrent chain).
      v16h va = loadA(inb + (tt * CHK_ + nCol) * HP_, hiH);
      v8f xp_r[2], xp_z[2], xp_n[2];
      xp_r[0] = wmma16x16x32(va, Bi[0], splat8(brz[0]));
      xp_r[1] = wmma16x16x32(va, Bi[1], splat8(brz[1]));
      xp_z[0] = wmma16x16x32(va, Bi[2], splat8(brz[2]));
      xp_z[1] = wmma16x16x32(va, Bi[3], splat8(brz[3]));
      xp_n[0] = wmma16x16x32(va, Bi[4], splat8(bxn[0]));
      xp_n[1] = wmma16x16x32(va, Bi[5], splat8(bxn[1]));

      // Recurrent matmul: h (A-layout from scratch) x Whh, C = x-projection.
      v16h ha = loadA(hs + nCol * HP_, hiH);
      v8f a_r[2], a_z[2], a_hn[2];
      a_r[0]  = wmma16x16x32(ha, Bh[0], xp_r[0]);
      a_r[1]  = wmma16x16x32(ha, Bh[1], xp_r[1]);
      a_z[0]  = wmma16x16x32(ha, Bh[2], xp_z[0]);
      a_z[1]  = wmma16x16x32(ha, Bh[3], xp_z[1]);
      a_hn[0] = wmma16x16x32(ha, Bh[4], splat8(bhn[0]));
      a_hn[1] = wmma16x16x32(ha, Bh[5], splat8(bhn[1]));

      // Activations: all gates for (b,j) live in the same lane/register slot.
#pragma unroll
      for (int p = 0; p < 2; ++p) {
#pragma unroll
        for (int r = 0; r < 8; ++r) {
          const float rr = sigm(a_r[p][r]);
          const float zz = sigm(a_z[p][r]);
          const float nn = tanhfast(xp_n[p][r] + rr * a_hn[p][r]);
          hst[p][r] = nn + zz * (hst[p][r] - nn);   // (1-z)*n + z*h
        }
      }

      // Write back: columns j and j+16 pack into one dword at permuted K slots
      // 2j / 2j+1 -> one v_cvt_pk_rtz + one ds_store_b32 per row per target.
#pragma unroll
      for (int r = 0; r < 8; ++r) {
        const int brow = r + 8 * hiH;
        union { v2fp16 h2; uint32_t u; } pk;
        pk.h2 = __builtin_amdgcn_cvt_pkrtz(hst[0][r], hst[1][r]);
        *(uint32_t*)(hs + brow * HP_ + 2 * nCol) = pk.u;
        if (outb)
          *(uint32_t*)(outb + (tt * CHK_ + brow) * HP_ + 2 * nCol) = pk.u;
      }
      if (last_layer) {
#pragma unroll
        for (int p = 0; p < 2; ++p) {
          const int col = p * 16 + nCol;
          if (col < H_) {
#pragma unroll
            for (int r = 0; r < 8; ++r) {
              const int brow = r + 8 * hiH;
              P.out_h[((size_t)(b0 + brow) * T_ + t) * H_ + col] = hst[p][r];
            }
          }
        }
      }
    }

    if (mt == NT_ - 1) {                 // final hidden state of this layer
#pragma unroll
      for (int p = 0; p < 2; ++p) {
        const int col = p * 16 + nCol;
        if (col < H_) {
#pragma unroll
          for (int r = 0; r < 8; ++r) {
            const int brow = r + 8 * hiH;
            P.out_hn[((size_t)(b0 + brow) * L_ + wave) * H_ + col] = hst[p][r];
          }
        }
      }
    }
  }
}

// y_hat = h_n.reshape(B, L*H) @ W_out.T + b_out  (tiny: 256 x 100 dot products)
__global__ void gru_head(const float* __restrict__ hn,
                         const float* __restrict__ Wout,
                         const float* __restrict__ bout,
                         float* __restrict__ y) {
  const int b = blockIdx.x * blockDim.x + threadIdx.x;
  if (b < B_) {
    float acc = bout[0];
#pragma unroll 4
    for (int i = 0; i < L_ * H_; ++i) acc += hn[b * L_ * H_ + i] * Wout[i];
    y[b] = acc;
  }
}

extern "C" void kernel_launch(void* const* d_in, const int* in_sizes, int n_in,
                              void* d_out, int out_size, void* d_ws,
                              size_t ws_size, hipStream_t stream) {
  (void)in_sizes; (void)n_in; (void)out_size; (void)d_ws; (void)ws_size;

  GruParams P;
  P.x = (const float*)d_in[0];
  for (int l = 0; l < L_; ++l) {
    P.Wih[l] = (const float*)d_in[1 + 4 * l];
    P.Whh[l] = (const float*)d_in[2 + 4 * l];
    P.bih[l] = (const float*)d_in[3 + 4 * l];
    P.bhh[l] = (const float*)d_in[4 + 4 * l];
  }
  float* out = (float*)d_out;
  P.out_h  = out;                                      // [B,T,H]
  P.out_hn = out + (size_t)B_ * T_ * H_;               // [B,L,H]
  float* y = out + (size_t)B_ * T_ * H_ + (size_t)B_ * L_ * H_;  // [B,1]

  gru_pipeline<<<B_ / CHK_, L_ * 32, 0, stream>>>(P);
  gru_head<<<1, B_, 0, stream>>>(P.out_hn, (const float*)d_in[21],
                                 (const float*)d_in[22], y);
}
